// BasicTransformerBlock_33689723470467
// MI455X (gfx1250) — compile-verified, matrix operand order checked by reading, not run on hardware
//
#include <hip/hip_runtime.h>
#include <hip/hip_bf16.h>
#include <math.h>

// ---------------- problem constants ----------------
#define HID   1280
#define CROSS 2048
#define HEAD  64
#define NH    (HID / HEAD)      // 20
#define FF    (4 * HID)         // 5120
#define TQ    2048
#define BATCH 2
#define MTOK  (BATCH * TQ)      // 4096
#define SKV   77
#define MKV   (BATCH * SKV)     // 154

typedef __attribute__((ext_vector_type(16))) _Float16 v16h;
typedef __attribute__((ext_vector_type(8)))  float    v8f;
typedef unsigned __attribute__((ext_vector_type(4))) v4u;
typedef int      __attribute__((ext_vector_type(4))) v4i;
typedef int      __attribute__((ext_vector_type(8))) v8i;

union FragU { v16h h; unsigned u[8]; };
union H16U  { _Float16 h; unsigned short u; };

// ---------------- Tensor Data Mover (CDNA5) ----------------
#if defined(__HIP_DEVICE_COMPILE__) && __has_builtin(__builtin_amdgcn_tensor_load_to_lds)
#define HAVE_TDM 1
#if __clang_major__ >= 23
#define TDM_SIX_ARG 1
#endif
#endif

#if defined(HAVE_TDM)
// Issue one TDM 2D-tile load: tile_d0 contiguous elements (f16) per row,
// tile_d1 rows, global row stride `row_stride` elements. LDS destination gets
// `pad_amount_dw` DWORDs of padding after every `pad_interval_dw` DWORDs
// (encoded per D# group1). OOB rows (y >= tensor_d1) read as zero.
__device__ __forceinline__ void tdm_load_2d(const _Float16* gsrc, unsigned lds_byte,
                                            unsigned tile_d0, unsigned tile_d1,
                                            unsigned tensor_d1, unsigned row_stride,
                                            unsigned pad_int_enc, unsigned pad_amt_enc) {
  unsigned long long ga = (unsigned long long)(const void*)gsrc;
  v4u g0;
  g0[0] = 1u;                                            // count=1, user descriptor
  g0[1] = lds_byte;                                      // lds_addr (bytes)
  g0[2] = (unsigned)(ga & 0xFFFFFFFFu);                  // global_addr[31:0]
  g0[3] = (unsigned)((ga >> 32) & 0x01FFFFFFu) | (2u << 30);  // addr[56:32], type=2
  const unsigned td0 = 0x40000000u;                      // huge dim0: no X clipping
  v8i g1;
  g1[0] = (int)((1u << 16)            // data_size = 2 bytes
              | (1u << 20)            // pad_enable
              | (pad_int_enc << 22)   // pad interval (2<<enc DWORDs)
              | (pad_amt_enc << 25)); // pad amount (enc+1 DWORDs)
  g1[1] = (int)((td0 & 0xFFFFu) << 16);                            // td0[15:0]
  g1[2] = (int)((td0 >> 16) | ((tensor_d1 & 0xFFFFu) << 16));      // td0 hi, td1 lo
  g1[3] = (int)((tensor_d1 >> 16) | (tile_d0 << 16));              // td1 hi, tile_dim0
  g1[4] = (int)(tile_d1 & 0xFFFFu);                                // tile_dim1, tile_dim2=0
  g1[5] = (int)row_stride;                                         // dim0_stride[31:0]
  g1[6] = 0;                                                       // stride0 hi, stride1 lo
  g1[7] = 0;
  v4i z4 = {0, 0, 0, 0};
#if defined(TDM_SIX_ARG)
  v8i z8 = {0, 0, 0, 0, 0, 0, 0, 0};
  __builtin_amdgcn_tensor_load_to_lds(g0, g1, z4, z4, z8, 0);
#else
  __builtin_amdgcn_tensor_load_to_lds(g0, g1, z4, z4, 0);
#endif
}
__device__ __forceinline__ void tdm_wait() { __builtin_amdgcn_s_wait_tensorcnt(0); }
// Flat LDS pointer low 32 bits == wave-relative LDS byte address (ISA 10.2).
__device__ __forceinline__ unsigned lds_byte_off(const void* p) {
  return (unsigned)(unsigned long long)p;
}
#endif

// Gather a 16x32 f16 WMMA A/B fragment from a row-major half tile.
// Per ISA layout: lane L owns row (L&15); lanes 0-15 take K={0..7,16..23},
// lanes 16-31 take K={8..15,24..31}; each VGPR packs 2 consecutive K halves.
__device__ __forceinline__ v16h frag_gather(const unsigned short* __restrict__ base,
                                            int row, int stride, int lane) {
  const int kh = (lane & 16) ? 8 : 0;
  const unsigned short* p = base + (size_t)row * stride;
  FragU f;
#pragma unroll
  for (int j = 0; j < 4; ++j) {
    f.u[j]     = *(const unsigned*)(p + kh + 2 * j);
    f.u[4 + j] = *(const unsigned*)(p + 16 + kh + 2 * j);
  }
  return f.h;
}

__device__ __forceinline__ float hmax16(float v) {
#pragma unroll
  for (int m = 1; m < 16; m <<= 1) v = fmaxf(v, __shfl_xor(v, m, 32));
  return v;
}
__device__ __forceinline__ float hsum16(float v) {
#pragma unroll
  for (int m = 1; m < 16; m <<= 1) v += __shfl_xor(v, m, 32);
  return v;
}

// ---------------- f32 -> f16 convert ----------------
__global__ void cvt_f32_f16(const float* __restrict__ in, _Float16* __restrict__ out, size_t n) {
  size_t i = (size_t)blockIdx.x * blockDim.x + threadIdx.x;
  size_t stride = (size_t)gridDim.x * blockDim.x;
  for (; i < n; i += stride) out[i] = (_Float16)in[i];
}

// ---------------- LayerNorm -> f16 ----------------
__global__ __launch_bounds__(256)
void ln_f16(const float* __restrict__ x, const float* __restrict__ g,
            const float* __restrict__ bt, _Float16* __restrict__ out) {
  const int row = blockIdx.x;
  const float* xr = x + (size_t)row * HID;
  float vals[HID / 256];
  float s = 0.f, s2 = 0.f;
#pragma unroll
  for (int i = 0; i < HID / 256; ++i) {
    float v = xr[threadIdx.x + i * 256];
    vals[i] = v; s += v; s2 += v * v;
  }
  __shared__ float rs[256], rs2[256];
  rs[threadIdx.x] = s; rs2[threadIdx.x] = s2;
  __syncthreads();
  for (int off = 128; off > 0; off >>= 1) {
    if (threadIdx.x < off) { rs[threadIdx.x] += rs[threadIdx.x + off]; rs2[threadIdx.x] += rs2[threadIdx.x + off]; }
    __syncthreads();
  }
  const float mu = rs[0] * (1.0f / HID);
  const float var = rs2[0] * (1.0f / HID) - mu * mu;
  const float inv = rsqrtf(var + 1e-5f);
  _Float16* orow = out + (size_t)row * HID;
#pragma unroll
  for (int i = 0; i < HID / 256; ++i) {
    int c = threadIdx.x + i * 256;
    orow[c] = (_Float16)((vals[i] - mu) * inv * g[c] + bt[c]);
  }
}

// ---------------- tiled WMMA GEMM: C[M,N] = A[M,K] @ W[N,K]^T (+bias)(+residual) ----
// Block tile 128x64, BK=32, 8 waves as 4(M) x 2(N), wave tile 32x32 (2x2 WMMA).
// Weight tile staged by the Tensor Data Mover (wave 0), activations by vector loads.
template <bool OUT_F16>
__global__ __launch_bounds__(256)
void gemm_wmma_f16(const _Float16* __restrict__ A, const _Float16* __restrict__ W,
                   const float* __restrict__ bias, const float* __restrict__ resid,
                   void* __restrict__ outp, int M, int N, int K) {
  __shared__ __align__(16) unsigned short As[128 * 40];
  __shared__ __align__(16) unsigned short Bs[64 * 40];
  const int tid  = threadIdx.x;
  const int lane = tid & 31;
  const int wid  = tid >> 5;
  const int wm   = wid & 3;   // 0..3
  const int wn   = wid >> 2;  // 0..1
  const int bm0  = blockIdx.y * 128;
  const int bn0  = blockIdx.x * 64;

  v8f acc[2][2] = {};

  for (int k0 = 0; k0 < K; k0 += 32) {
    __syncthreads();
#if defined(HAVE_TDM)
    // 64x32 f16 weight tile, row stride K elems; pad 4 DWORDs after each
    // 16-DWORD (64 B) row -> LDS row stride 40 halves (matches frag_gather).
    if (wid == 0)
      tdm_load_2d(W + (size_t)bn0 * K + k0, lds_byte_off(Bs),
                  /*tile_d0=*/32, /*tile_d1=*/64, /*tensor_d1=*/64,
                  /*row_stride=*/(unsigned)K, /*pad_int_enc=*/3, /*pad_amt_enc=*/3);
#else
    {
      int r = tid >> 2, c = tid & 3;
      int gn = bn0 + r;
      uint4 val{0u, 0u, 0u, 0u};
      if (gn < N) val = *(const uint4*)(W + (size_t)gn * K + k0 + c * 8);
      *(uint4*)(&Bs[r * 40 + c * 8]) = val;
    }
#endif
    // stage A tile: 128 rows x 32 halves = 512 uint4
#pragma unroll
    for (int i = 0; i < 2; ++i) {
      int idx = tid + i * 256;
      int r = idx >> 2, c = idx & 3;
      int gm = bm0 + r;
      uint4 val{0u, 0u, 0u, 0u};
      if (gm < M) val = *(const uint4*)(A + (size_t)gm * K + k0 + c * 8);
      *(uint4*)(&As[r * 40 + c * 8]) = val;
    }
#if defined(HAVE_TDM)
    if (wid == 0) tdm_wait();
#endif
    __syncthreads();

    v16h af[2], bf[2];
#pragma unroll
    for (int mt = 0; mt < 2; ++mt)
      af[mt] = frag_gather(As, wm * 32 + mt * 16 + (lane & 15), 40, lane);
#pragma unroll
    for (int nt = 0; nt < 2; ++nt)
      bf[nt] = frag_gather(Bs, wn * 32 + nt * 16 + (lane & 15), 40, lane);
#pragma unroll
    for (int mt = 0; mt < 2; ++mt)
#pragma unroll
      for (int nt = 0; nt < 2; ++nt)
        acc[mt][nt] = __builtin_amdgcn_wmma_f32_16x16x32_f16(
            false, af[mt], false, bf[nt], (short)0, acc[mt][nt], false, false);
  }

  // store (C layout: VGPR j -> row j (lanes 0-15) / row j+8 (lanes 16-31))
  const int cm = (lane & 16) ? 8 : 0;
#pragma unroll
  for (int mt = 0; mt < 2; ++mt) {
#pragma unroll
    for (int nt = 0; nt < 2; ++nt) {
      const int gn = bn0 + wn * 32 + nt * 16 + (lane & 15);
      const float bv = bias ? bias[gn] : 0.0f;
#pragma unroll
      for (int j = 0; j < 8; ++j) {
        int gm = bm0 + wm * 32 + mt * 16 + j + cm;
        if (gm < M) {
          float v = acc[mt][nt][j] + bv;
          if (resid) v += resid[(size_t)gm * N + gn];
          if (OUT_F16) ((_Float16*)outp)[(size_t)gm * N + gn] = (_Float16)v;
          else         ((float*)outp)[(size_t)gm * N + gn] = v;
        }
      }
    }
  }
}

// ---------------- flash attention (online softmax, WMMA) ----------------
// grid: (T/128, B*NH), block 256 (8 waves, 16 q-rows each). 32-key tiles.
// K tile staged by the TDM (OOB rows zero-filled by tensor_dim clipping).
__global__ __launch_bounds__(256)
void flash_attn_wmma(const _Float16* __restrict__ q, const _Float16* __restrict__ k,
                     const _Float16* __restrict__ v, _Float16* __restrict__ o,
                     int T, int S) {
  __shared__ __align__(16) unsigned short Ks[32 * 72];      // (key, d) row-major
  __shared__ __align__(16) unsigned short Vt[64 * 40];      // (d, key) transposed
  __shared__ __align__(16) unsigned short Ps[8 * 16 * 36];  // per-wave P bounce

  const int tid  = threadIdx.x;
  const int lane = tid & 31;
  const int wid  = tid >> 5;
  const int bh = blockIdx.y;
  const int b  = bh / NH;
  const int h  = bh % NH;
  const int t0 = blockIdx.x * 128 + wid * 16;

  const _Float16* qb = q + ((size_t)b * T) * HID + h * HEAD;
  const _Float16* kb = k + ((size_t)b * S) * HID + h * HEAD;
  const _Float16* vb = v + ((size_t)b * S) * HID + h * HEAD;

  // Q fragments (A layout), d split into 2 chunks of 32
  v16h qf[2];
  {
    const unsigned short* qr =
        (const unsigned short*)(qb + (size_t)(t0 + (lane & 15)) * HID);
    const int kh = (lane & 16) ? 8 : 0;
#pragma unroll
    for (int c = 0; c < 2; ++c) {
      FragU f;
#pragma unroll
      for (int j = 0; j < 4; ++j) {
        f.u[j]     = *(const unsigned*)(qr + c * 32 + kh + 2 * j);
        f.u[4 + j] = *(const unsigned*)(qr + c * 32 + 16 + kh + 2 * j);
      }
      qf[c] = f.h;
    }
  }

  v8f oacc[4] = {};
  float mrun[8], lrun[8];
#pragma unroll
  for (int j = 0; j < 8; ++j) { mrun[j] = -1e30f; lrun[j] = 0.f; }

  const float scale = 0.125f;  // 1/sqrt(64)
  const int nkt = (S + 31) / 32;

  for (int kt = 0; kt < nkt; ++kt) {
    __syncthreads();
#if defined(HAVE_TDM)
    // 32x64 f16 K tile; pad 4 DWORDs after each 32-DWORD (128 B) row ->
    // LDS row stride 72 halves. tensor_d1 = S - kt*32 zero-fills ragged rows.
    if (wid == 0)
      tdm_load_2d(kb + (size_t)(kt * 32) * HID, lds_byte_off(Ks),
                  /*tile_d0=*/64, /*tile_d1=*/32,
                  /*tensor_d1=*/(unsigned)(S - kt * 32),
                  /*row_stride=*/HID, /*pad_int_enc=*/4, /*pad_amt_enc=*/3);
#else
    {
      int r = tid >> 3, c = tid & 7;
      int key = kt * 32 + r;
      uint4 val{0u, 0u, 0u, 0u};
      if (key < S) val = *(const uint4*)(kb + (size_t)key * HID + c * 8);
      *(uint4*)(&Ks[r * 72 + c * 8]) = val;
    }
#endif
    // stage V transposed (scalar): Vt[d][key]
#pragma unroll
    for (int i = 0; i < 8; ++i) {
      int e = tid + i * 256;
      int kk = e >> 6, d = e & 63;
      int key = kt * 32 + kk;
      unsigned short val = 0;
      if (key < S) val = *((const unsigned short*)vb + (size_t)key * HID + d);
      Vt[d * 40 + kk] = val;
    }
#if defined(HAVE_TDM)
    if (wid == 0) tdm_wait();
#endif
    __syncthreads();

    // S = Q @ K^T  (two 16-key sub-tiles, K-dim 64 = 2 WMMA each)
    v8f sc[2];
#pragma unroll
    for (int st = 0; st < 2; ++st) {
      v16h b0 = frag_gather(Ks,      st * 16 + (lane & 15), 72, lane);  // d 0..31
      v16h b1 = frag_gather(Ks + 32, st * 16 + (lane & 15), 72, lane);  // d 32..63
      v8f z = {};
      z = __builtin_amdgcn_wmma_f32_16x16x32_f16(false, qf[0], false, b0, (short)0, z, false, false);
      z = __builtin_amdgcn_wmma_f32_16x16x32_f16(false, qf[1], false, b1, (short)0, z, false, false);
      sc[st] = z;
    }
    const int key0 = kt * 32 + (lane & 15);
    const bool oob0 = (key0 >= S);
    const bool oob1 = (key0 + 16 >= S);
#pragma unroll
    for (int j = 0; j < 8; ++j) {
      sc[0][j] = oob0 ? -1e30f : sc[0][j] * scale;
      sc[1][j] = oob1 ? -1e30f : sc[1][j] * scale;
    }

    // online softmax update per owned row
    const int prow_off = (lane & 16) ? 8 : 0;
    unsigned short* pw = &Ps[wid * 16 * 36];
#pragma unroll
    for (int j = 0; j < 8; ++j) {
      float mx = hmax16(fmaxf(sc[0][j], sc[1][j]));
      float mn = fmaxf(mrun[j], mx);
      float alpha = __expf(mrun[j] - mn);
      mrun[j] = mn;
      lrun[j] *= alpha;
#pragma unroll
      for (int nt = 0; nt < 4; ++nt) oacc[nt][j] *= alpha;
      float p0 = __expf(sc[0][j] - mn);
      float p1 = __expf(sc[1][j] - mn);
      H16U h0, h1; h0.h = (_Float16)p0; h1.h = (_Float16)p1;
      pw[(j + prow_off) * 36 + (lane & 15)]      = h0.u;
      pw[(j + prow_off) * 36 + 16 + (lane & 15)] = h1.u;
      lrun[j] += hsum16(p0 + p1);
    }
    __syncthreads();  // make P visible across the wave's lanes via LDS

    // O += P @ V   (K-dim = 32 keys, 4 output 16-col tiles)
    v16h pf = frag_gather(pw, (lane & 15), 36, lane);
#pragma unroll
    for (int nt = 0; nt < 4; ++nt) {
      v16h vf = frag_gather(Vt, nt * 16 + (lane & 15), 40, lane);
      oacc[nt] = __builtin_amdgcn_wmma_f32_16x16x32_f16(
          false, pf, false, vf, (short)0, oacc[nt], false, false);
    }
  }

  // normalize and store
  const int cm = (lane & 16) ? 8 : 0;
  float linv[8];
#pragma unroll
  for (int j = 0; j < 8; ++j) linv[j] = 1.0f / lrun[j];
  _Float16* ob = o + ((size_t)b * T) * HID + h * HEAD;
#pragma unroll
  for (int nt = 0; nt < 4; ++nt) {
#pragma unroll
    for (int j = 0; j < 8; ++j) {
      int t = t0 + j + cm;
      ob[(size_t)t * HID + nt * 16 + (lane & 15)] = (_Float16)(oacc[nt][j] * linv[j]);
    }
  }
}

// ---------------- GEGLU elementwise ----------------
__global__ void geglu_kernel(const _Float16* __restrict__ proj, _Float16* __restrict__ out,
                             size_t total) {
  size_t i = (size_t)blockIdx.x * blockDim.x + threadIdx.x;
  size_t stride = (size_t)gridDim.x * blockDim.x;
  for (; i < total; i += stride) {
    size_t r = i / FF, c = i % FF;
    float a = (float)proj[r * (2 * FF) + c];
    float g = (float)proj[r * (2 * FF) + FF + c];
    float ge = 0.5f * g * (1.0f + erff(g * 0.70710678f));
    out[i] = (_Float16)(a * ge);
  }
}

// ---------------- host-side orchestration ----------------
extern "C" void kernel_launch(void* const* d_in, const int* in_sizes, int n_in,
                              void* d_out, int out_size, void* d_ws, size_t ws_size,
                              hipStream_t stream) {
  (void)in_sizes; (void)n_in; (void)out_size; (void)ws_size;
  const float* x      = (const float*)d_in[0];
  const float* enc    = (const float*)d_in[1];
  const float* ln1_g  = (const float*)d_in[2];
  const float* ln1_b  = (const float*)d_in[3];
  const float* ln2_g  = (const float*)d_in[4];
  const float* ln2_b  = (const float*)d_in[5];
  const float* ln3_g  = (const float*)d_in[6];
  const float* ln3_b  = (const float*)d_in[7];
  const float* wq1 = (const float*)d_in[8];
  const float* wk1 = (const float*)d_in[9];
  const float* wv1 = (const float*)d_in[10];
  const float* wo1 = (const float*)d_in[11];
  const float* bo1 = (const float*)d_in[12];
  const float* wq2 = (const float*)d_in[13];
  const float* wk2 = (const float*)d_in[14];
  const float* wv2 = (const float*)d_in[15];
  const float* wo2 = (const float*)d_in[16];
  const float* bo2 = (const float*)d_in[17];
  const float* w_geglu = (const float*)d_in[18];
  const float* b_geglu = (const float*)d_in[19];
  const float* w_ffout = (const float*)d_in[20];
  const float* b_ffout = (const float*)d_in[21];

  // bump allocator over d_ws
  char* ws = (char*)d_ws;
  size_t off = 0;
  auto alloc = [&](size_t bytes) -> void* {
    void* p = ws + off;
    off = (off + bytes + 255) & ~(size_t)255;
    return p;
  };
  const size_t H2 = (size_t)HID * HID;
  const size_t HC = (size_t)HID * CROSS;
  _Float16* hwq1 = (_Float16*)alloc(H2 * 2);
  _Float16* hwk1 = (_Float16*)alloc(H2 * 2);
  _Float16* hwv1 = (_Float16*)alloc(H2 * 2);
  _Float16* hwo1 = (_Float16*)alloc(H2 * 2);
  _Float16* hwq2 = (_Float16*)alloc(H2 * 2);
  _Float16* hwk2 = (_Float16*)alloc(HC * 2);
  _Float16* hwv2 = (_Float16*)alloc(HC * 2);
  _Float16* hwo2 = (_Float16*)alloc(H2 * 2);
  _Float16* hwgg = (_Float16*)alloc((size_t)2 * FF * HID * 2);
  _Float16* hwff = (_Float16*)alloc((size_t)HID * FF * 2);
  _Float16* enc16 = (_Float16*)alloc((size_t)MKV * CROSS * 2);
  _Float16* h16   = (_Float16*)alloc((size_t)MTOK * HID * 2);
  _Float16* q16   = (_Float16*)alloc((size_t)MTOK * HID * 2);
  _Float16* k16   = (_Float16*)alloc((size_t)MTOK * HID * 2);
  _Float16* v16   = (_Float16*)alloc((size_t)MTOK * HID * 2);
  _Float16* a16   = (_Float16*)alloc((size_t)MTOK * HID * 2);
  float*    xa    = (float*)alloc((size_t)MTOK * HID * 4);
  float*    xb    = (float*)alloc((size_t)MTOK * HID * 4);
  _Float16* proj16 = (_Float16*)alloc((size_t)MTOK * 2 * FF * 2);
  _Float16* ffin16 = (_Float16*)alloc((size_t)MTOK * FF * 2);

  auto cvt = [&](const float* src, _Float16* dst, size_t n) {
    int blocks = (int)((n + 2047) / 2048);
    if (blocks > 4096) blocks = 4096;
    cvt_f32_f16<<<blocks, 256, 0, stream>>>(src, dst, n);
  };
  auto gemm = [&](const _Float16* A, const _Float16* W, const float* bias,
                  const float* resid, void* out, int M, int N, int K, bool out16) {
    dim3 grid(N / 64, (M + 127) / 128);
    if (out16) gemm_wmma_f16<true ><<<grid, 256, 0, stream>>>(A, W, bias, resid, out, M, N, K);
    else       gemm_wmma_f16<false><<<grid, 256, 0, stream>>>(A, W, bias, resid, out, M, N, K);
  };

  // weights + encoder to f16
  cvt(wq1, hwq1, H2); cvt(wk1, hwk1, H2); cvt(wv1, hwv1, H2); cvt(wo1, hwo1, H2);
  cvt(wq2, hwq2, H2); cvt(wk2, hwk2, HC); cvt(wv2, hwv2, HC); cvt(wo2, hwo2, H2);
  cvt(w_geglu, hwgg, (size_t)2 * FF * HID);
  cvt(w_ffout, hwff, (size_t)HID * FF);
  cvt(enc, enc16, (size_t)MKV * CROSS);

  // ---- self attention ----
  ln_f16<<<MTOK, 256, 0, stream>>>(x, ln1_g, ln1_b, h16);
  gemm(h16, hwq1, nullptr, nullptr, q16, MTOK, HID, HID, true);
  gemm(h16, hwk1, nullptr, nullptr, k16, MTOK, HID, HID, true);
  gemm(h16, hwv1, nullptr, nullptr, v16, MTOK, HID, HID, true);
  flash_attn_wmma<<<dim3(TQ / 128, BATCH * NH), 256, 0, stream>>>(q16, k16, v16, a16, TQ, TQ);
  gemm(a16, hwo1, bo1, x, xa, MTOK, HID, HID, false);

  // ---- cross attention ----
  ln_f16<<<MTOK, 256, 0, stream>>>(xa, ln2_g, ln2_b, h16);
  gemm(h16, hwq2, nullptr, nullptr, q16, MTOK, HID, HID, true);
  gemm(enc16, hwk2, nullptr, nullptr, k16, MKV, HID, CROSS, true);
  gemm(enc16, hwv2, nullptr, nullptr, v16, MKV, HID, CROSS, true);
  flash_attn_wmma<<<dim3(TQ / 128, BATCH * NH), 256, 0, stream>>>(q16, k16, v16, a16, TQ, SKV);
  gemm(a16, hwo2, bo2, xa, xb, MTOK, HID, HID, false);

  // ---- GEGLU feed-forward ----
  ln_f16<<<MTOK, 256, 0, stream>>>(xb, ln3_g, ln3_b, h16);
  gemm(h16, hwgg, b_geglu, nullptr, proj16, MTOK, 2 * FF, HID, true);
  geglu_kernel<<<4096, 256, 0, stream>>>(proj16, ffin16, (size_t)MTOK * FF);
  gemm(ffin16, hwff, b_ffout, xb, d_out, MTOK, HID, FF, false);
}
